// Pointnet2MSG_12678743458428
// MI455X (gfx1250) — matrix operand order repaired and checked by reading, not compiled, not run on hardware
//
#include <hip/hip_runtime.h>
#include <hip/hip_bf16.h>

typedef __attribute__((ext_vector_type(16))) _Float16 v16h;
typedef __attribute__((ext_vector_type(8)))  _Float16 v8h;
typedef __attribute__((ext_vector_type(8)))  float    v8f;

namespace {

constexpr int kB  = 2;
constexpr int kN  = 16384;
constexpr int kS  = 4096;
constexpr int kCF = 6;

__device__ __forceinline__ void amax2(float& v, int& i, float ov, int oi) {
  if (ov > v || (ov == v && oi < i)) { v = ov; i = oi; }
}

// ---------------------------------------------------------------- FPS -------
// One workgroup per batch. Points + running min-distances live in registers.
__global__ __launch_bounds__(1024) void fps_kernel(
    const float* __restrict__ xyz, int* __restrict__ fpsIdx,
    float* __restrict__ newXyz) {
  const int b = blockIdx.x;
  const int t = threadIdx.x;
  constexpr int NT  = 1024;
  constexpr int PPT = kN / NT;  // 16 points per thread
  __shared__ float redD[32];
  __shared__ int   redI[32];
  __shared__ float bc[3];
  __shared__ int   sel;
  float px[PPT], py[PPT], pz[PPT], dist[PPT];
  const float* X = xyz + (size_t)b * kN * 3;
#pragma unroll
  for (int i = 0; i < PPT; ++i) {
    int p = t + i * NT;
    px[i] = X[p * 3 + 0]; py[i] = X[p * 3 + 1]; pz[i] = X[p * 3 + 2];
    dist[i] = 1e10f;
  }
  if (t == 0) {
    fpsIdx[(size_t)b * kS] = 0;
    bc[0] = X[0]; bc[1] = X[1]; bc[2] = X[2];
    newXyz[(size_t)b * kS * 3 + 0] = X[0];
    newXyz[(size_t)b * kS * 3 + 1] = X[1];
    newXyz[(size_t)b * kS * 3 + 2] = X[2];
  }
  __syncthreads();
  const int wid = t >> 5, lane = t & 31;
  for (int j = 1; j < kS; ++j) {
    float lx = bc[0], ly = bc[1], lz = bc[2];
    float bv = -1.0f; int bi = 0;
#pragma unroll
    for (int i = 0; i < PPT; ++i) {
      float dx = px[i] - lx, dy = py[i] - ly, dz = pz[i] - lz;
      float d = dx * dx + dy * dy + dz * dz;
      float nd = fminf(dist[i], d);
      dist[i] = nd;
      if (nd > bv) { bv = nd; bi = t + i * NT; }
    }
#pragma unroll
    for (int off = 16; off > 0; off >>= 1) {
      float ov = __shfl_down(bv, off, 32);
      int   oi = __shfl_down(bi, off, 32);
      amax2(bv, bi, ov, oi);
    }
    if (lane == 0) { redD[wid] = bv; redI[wid] = bi; }
    __syncthreads();
    if (wid == 0) {
      bv = redD[lane]; bi = redI[lane];
#pragma unroll
      for (int off = 16; off > 0; off >>= 1) {
        float ov = __shfl_down(bv, off, 32);
        int   oi = __shfl_down(bi, off, 32);
        amax2(bv, bi, ov, oi);
      }
      if (lane == 0) { sel = bi; fpsIdx[(size_t)b * kS + j] = bi; }
    }
    __syncthreads();
    int bsel = sel;
    if ((bsel & (NT - 1)) == t) {
      int i = bsel >> 10;
      bc[0] = px[i]; bc[1] = py[i]; bc[2] = pz[i];
      newXyz[((size_t)b * kS + j) * 3 + 0] = px[i];
      newXyz[((size_t)b * kS + j) * 3 + 1] = py[i];
      newXyz[((size_t)b * kS + j) * 3 + 2] = pz[i];
    }
    __syncthreads();
  }
}

// ------------------------------------------------------------ ball query ----
// One wave per query; ordered compaction of first nsample in-radius indices.
__global__ void ball_query_kernel(const float* __restrict__ xyz,
                                  const float* __restrict__ newXyz,
                                  int* __restrict__ idx, int nsample, float r2) {
  int wave = (blockIdx.x * blockDim.x + threadIdx.x) >> 5;
  int lane = threadIdx.x & 31;
  if (wave >= kB * kS) return;
  int b = wave / kS;
  float cx = newXyz[(size_t)wave * 3 + 0];
  float cy = newXyz[(size_t)wave * 3 + 1];
  float cz = newXyz[(size_t)wave * 3 + 2];
  const float* X = xyz + (size_t)b * kN * 3;
  int cnt = 0, firstIdx = kN - 1;
  bool haveFirst = false;
  for (int base = 0; base < kN && cnt < nsample; base += 32) {
    int p = base + lane;
    float dx = X[p * 3 + 0] - cx, dy = X[p * 3 + 1] - cy, dz = X[p * 3 + 2] - cz;
    float d2 = dx * dx + dy * dy + dz * dz;
    bool inr = d2 < r2;
    unsigned m = (unsigned)__ballot(inr);
    if (!haveFirst && m) { firstIdx = base + __ffs(m) - 1; haveFirst = true; }
    int pre = __popc(m & ((1u << lane) - 1u));
    if (inr && (cnt + pre) < nsample) idx[(size_t)wave * nsample + cnt + pre] = p;
    cnt += __popc(m);
  }
  if (cnt < nsample && lane >= cnt && lane < nsample)
    idx[(size_t)wave * nsample + lane] = firstIdx;
}

// ------------------------------------------------------------- grouping -----
// Build grouped input f16 [ (b,s,k) ][ 32 ]: 3 centered xyz + 6 feats + zeros.
__global__ void group_kernel(const float* __restrict__ xyz,
                             const float* __restrict__ newXyz,
                             const float* __restrict__ feat,
                             const int* __restrict__ idx,
                             _Float16* __restrict__ h0, int K) {
  int t = blockIdx.x * blockDim.x + threadIdx.x;
  int total = kB * kS * K;
  if (t >= total) return;
  int sg = t / K;
  int b = sg / kS;
  int p = idx[t];
  const float* P = xyz + ((size_t)b * kN + p) * 3;
  const float* C = newXyz + (size_t)sg * 3;
  _Float16* o = h0 + (size_t)t * 32;
  o[0] = (_Float16)(P[0] - C[0]);
  o[1] = (_Float16)(P[1] - C[1]);
  o[2] = (_Float16)(P[2] - C[2]);
#pragma unroll
  for (int c = 0; c < kCF; ++c)
    o[3 + c] = (_Float16)feat[((size_t)b * kCF + c) * kN + p];
#pragma unroll
  for (int c = 9; c < 32; ++c) o[c] = (_Float16)0.f;
}

// ------------------------------------------------------- weight prep --------
__global__ void prep_w_kernel(const float* __restrict__ src,
                              _Float16* __restrict__ dst,
                              int O, int C, int Opad, int Cpad) {
  int t = blockIdx.x * blockDim.x + threadIdx.x;
  if (t >= Opad * Cpad) return;
  int c = t % Cpad, o = t / Cpad;
  dst[t] = (o < O && c < C) ? (_Float16)src[(size_t)o * C + c] : (_Float16)0.f;
}

__global__ void img_transpose_kernel(const float* __restrict__ src,
                                     _Float16* __restrict__ dst) {
  int t = blockIdx.x * blockDim.x + threadIdx.x;
  int total = kB * kS * 64;
  if (t >= total) return;
  int c = t & 63;
  int sg = t >> 6;
  int b = sg / kS, s = sg % kS;
  dst[t] = (_Float16)src[((size_t)b * 64 + c) * kS + s];
}

// ------------------------------------------------- WMMA conv1d + bias + act -
// out[p][o] = act( sum_c W[o][c] * in[p][c] + bias[o] )
// in: f16 [Ptot][ldin], W: f16 [Opad][Kt*32] zero-padded, out f16 or f32.
__global__ void wmma_conv_kernel(const _Float16* __restrict__ in,
                                 const _Float16* __restrict__ Wp,
                                 const float* __restrict__ bias,
                                 _Float16* __restrict__ out16,
                                 float* __restrict__ out32,
                                 int Ptot, int ldin, int Kt, int Mtiles, int O,
                                 int ldout, int chOff, int zeroTiles, int act,
                                 int PperB) {
  int wave = (blockIdx.x * blockDim.x + threadIdx.x) >> 5;
  int lane = threadIdx.x & 31;
  int nWork = (Ptot >> 4) * Mtiles;
  if (wave >= nWork) return;          // wave-uniform: EXEC stays all-ones
  int mt = wave % Mtiles;
  int nt = wave / Mtiles;
  int col = lane & 15;
  int hi  = lane >> 4;                // 0 or 1
  int p   = (nt << 4) + col;
  int Cpad = Kt << 5;
  int arow = mt * 16 + col;
  v8f acc = {};
  for (int kt = 0; kt < Kt; ++kt) {
    // A frag (16x32 f16): lanes 0-15 hold K 0..7 / 16..23, lanes 16-31 hold 8..15 / 24..31
    const _Float16* wrow = Wp + (size_t)arow * Cpad + (kt << 5);
    v8h alo = *(const v8h*)(wrow + hi * 8);
    v8h ahi = *(const v8h*)(wrow + 16 + hi * 8);
    v16h a;
#pragma unroll
    for (int i = 0; i < 8; ++i) { a[i] = alo[i]; a[8 + i] = ahi[i]; }
    // B frag (32x16 f16): lane = col, lanes 0-15 hold K 0..7 / 16..23, lanes 16-31 hold 8..15 / 24..31
    const _Float16* brow = in + (size_t)p * ldin + (kt << 5);
    v8h blo = *(const v8h*)(brow + hi * 8);
    v8h bhi = *(const v8h*)(brow + 16 + hi * 8);
    v16h bfr;
#pragma unroll
    for (int i = 0; i < 8; ++i) { bfr[i] = blo[i]; bfr[8 + i] = bhi[i]; }
    acc = __builtin_amdgcn_wmma_f32_16x16x32_f16(false, a, false, bfr,
                                                 (short)0, acc, false, false);
  }
  int chBase = mt * 16 + hi * 8;      // D layout: lane holds 8 consecutive rows
  float r[8];
#pragma unroll
  for (int v = 0; v < 8; ++v) {
    int ch = chBase + v;
    float x = acc[v] + (ch < O ? bias[ch] : 0.0f);
    if (act == 1) x = fmaxf(x, 0.0f);
    else if (act == 2) x = tanhf(x);
    r[v] = x;
  }
  if (out16) {
    v8h sv;
#pragma unroll
    for (int v = 0; v < 8; ++v) sv[v] = (_Float16)r[v];
    *(v8h*)(out16 + (size_t)p * ldout + chOff + chBase) = sv;
    if (mt == 0) {  // zero the K-padding channels for the next layer
      for (int zt = Mtiles; zt < Mtiles + zeroTiles; ++zt) {
        v8h z = {};
        *(v8h*)(out16 + (size_t)p * ldout + chOff + zt * 16 + hi * 8) = z;
      }
    }
  } else {
    int bb = p / PperB, s = p % PperB;
#pragma unroll
    for (int v = 0; v < 8; ++v) {
      int ch = chBase + v;
      if (ch < O) out32[((size_t)bb * O + ch) * PperB + s] = r[v];
    }
  }
}

// ------------------------------------------------------------- max pool -----
__global__ void pool_kernel(const _Float16* __restrict__ h,
                            _Float16* __restrict__ fuseIn,
                            int K, int O, int ldin, int chOff) {
  int t = blockIdx.x * blockDim.x + threadIdx.x;
  int total = kB * kS * O;
  if (t >= total) return;
  int o = t % O, sg = t / O;
  const _Float16* base = h + (size_t)sg * K * ldin + o;
  float m = (float)base[0];
  for (int k = 1; k < K; ++k) m = fmaxf(m, (float)base[(size_t)k * ldin]);
  fuseIn[(size_t)sg * 192 + chOff + o] = (_Float16)m;
}

// -------------------------------------------- attention (fc3+sigmoid) + mul -
__global__ void att_kernel(const _Float16* __restrict__ ri,
                           const float* __restrict__ w3,
                           const float* __restrict__ b3,
                           const _Float16* __restrict__ img,
                           _Float16* __restrict__ imgr) {
  int t = blockIdx.x * blockDim.x + threadIdx.x;
  if (t >= kB * kS) return;
  float s = b3[0];
#pragma unroll
  for (int c = 0; c < 24; ++c) s += (float)ri[(size_t)t * 32 + c] * w3[c];
  float att = 1.0f / (1.0f + expf(-s));
  float g = 1.0f + att;
  for (int c = 0; c < 96; ++c)
    imgr[(size_t)t * 96 + c] = (_Float16)((float)img[(size_t)t * 96 + c] * g);
}

}  // namespace

extern "C" void kernel_launch(void* const* d_in, const int* in_sizes, int n_in,
                              void* d_out, int out_size, void* d_ws, size_t ws_size,
                              hipStream_t stream) {
  (void)in_sizes; (void)n_in; (void)out_size; (void)ws_size;
  const float* xyz  = (const float*)d_in[0];
  const float* feat = (const float*)d_in[1];
  const float* imgF = (const float*)d_in[2];
  const float* w00 = (const float*)d_in[3];   const float* b00 = (const float*)d_in[4];
  const float* w01 = (const float*)d_in[5];   const float* b01 = (const float*)d_in[6];
  const float* w02 = (const float*)d_in[7];   const float* b02 = (const float*)d_in[8];
  const float* w10 = (const float*)d_in[9];   const float* b10 = (const float*)d_in[10];
  const float* w11 = (const float*)d_in[11];  const float* b11 = (const float*)d_in[12];
  const float* w12 = (const float*)d_in[13];  const float* b12 = (const float*)d_in[14];
  const float* wImg = (const float*)d_in[15]; const float* bImg = (const float*)d_in[16];
  const float* wFc2 = (const float*)d_in[17]; const float* bFc2 = (const float*)d_in[18];
  const float* wFc3 = (const float*)d_in[19]; const float* bFc3 = (const float*)d_in[20];
  const float* wPc  = (const float*)d_in[21]; const float* bPc  = (const float*)d_in[22];
  const float* wFuse = (const float*)d_in[23]; const float* bFuse = (const float*)d_in[24];
  float* out = (float*)d_out;

  char* ws = (char*)d_ws;
  size_t off = 0;
  auto alloc = [&](size_t bytes) -> char* {
    char* p = ws + off;
    off = (off + bytes + 255) & ~(size_t)255;
    return p;
  };

  // 64 MB ping-pong arena, reused by both scales (stream-serialized).
  char* arena = alloc(64ull << 20);
  _Float16* h0s0 = (_Float16*)(arena + 0);              // 8 MB  (131072 x 32 f16)
  _Float16* a0f  = (_Float16*)(arena + (8ull << 20));   // 8 MB
  _Float16* b0f  = (_Float16*)(arena + (16ull << 20));  // 8 MB
  _Float16* h0s1 = (_Float16*)(arena + 0);              // 16 MB (262144 x 32 f16)
  _Float16* a1f  = (_Float16*)(arena + (32ull << 20));  // 16 MB
  _Float16* b1f  = (_Float16*)(arena + (48ull << 20));  // 16 MB
  _Float16* out1 = (_Float16*)(arena + 0);              // 32 MB (262144 x 64 f16)

  int*   fpsIdx = (int*)alloc((size_t)kB * kS * 4);
  float* newXyz = (float*)alloc((size_t)kB * kS * 3 * 4);
  int*   idx0   = (int*)alloc((size_t)kB * kS * 16 * 4);
  int*   idx1   = (int*)alloc((size_t)kB * kS * 32 * 4);
  _Float16* imgT   = (_Float16*)alloc((size_t)kB * kS * 64 * 2);
  _Float16* img    = (_Float16*)alloc((size_t)kB * kS * 96 * 2);
  _Float16* ri     = (_Float16*)alloc((size_t)kB * kS * 32 * 2);
  _Float16* imgr   = (_Float16*)alloc((size_t)kB * kS * 96 * 2);
  _Float16* fuseIn = (_Float16*)alloc((size_t)kB * kS * 192 * 2);
  _Float16* Wp00 = (_Float16*)alloc(16 * 32 * 2);
  _Float16* Wp01 = (_Float16*)alloc(16 * 32 * 2);
  _Float16* Wp02 = (_Float16*)alloc(32 * 32 * 2);
  _Float16* Wp10 = (_Float16*)alloc(32 * 32 * 2);
  _Float16* Wp11 = (_Float16*)alloc(32 * 32 * 2);
  _Float16* Wp12 = (_Float16*)alloc(64 * 32 * 2);
  _Float16* WpImg  = (_Float16*)alloc(96 * 64 * 2);
  _Float16* WpFc2  = (_Float16*)alloc(32 * 96 * 2);
  _Float16* WpPc   = (_Float16*)alloc(96 * 96 * 2);
  _Float16* WpFuse = (_Float16*)alloc(128 * 192 * 2);

  auto cdiv = [](long long a, long long b) { return (int)((a + b - 1) / b); };
  auto prep = [&](const float* src, _Float16* dst, int O, int C, int Opad, int Cpad) {
    int n = Opad * Cpad;
    prep_w_kernel<<<cdiv(n, 256), 256, 0, stream>>>(src, dst, O, C, Opad, Cpad);
  };
  auto conv = [&](const _Float16* in, const _Float16* Wp, const float* bias,
                  _Float16* o16, float* o32, int Ptot, int ldin, int Kt, int Mt,
                  int O, int ldout, int chOff, int zt, int act, int PperB) {
    int nWork = (Ptot >> 4) * Mt;
    wmma_conv_kernel<<<cdiv(nWork, 8), 256, 0, stream>>>(
        in, Wp, bias, o16, o32, Ptot, ldin, Kt, Mt, O, ldout, chOff, zt, act, PperB);
  };

  // Weight conversion (f32 -> zero-padded f16)
  prep(w00, Wp00, 16, 9, 16, 32);
  prep(w01, Wp01, 16, 16, 16, 32);
  prep(w02, Wp02, 32, 16, 32, 32);
  prep(w10, Wp10, 32, 9, 32, 32);
  prep(w11, Wp11, 32, 32, 32, 32);
  prep(w12, Wp12, 64, 32, 64, 32);
  prep(wImg, WpImg, 96, 64, 96, 64);
  prep(wFc2, WpFc2, 24, 96, 32, 96);
  prep(wPc, WpPc, 96, 96, 96, 96);
  prep(wFuse, WpFuse, 128, 192, 128, 192);

  // FPS + image branch prep
  fps_kernel<<<kB, 1024, 0, stream>>>(xyz, fpsIdx, newXyz);
  img_transpose_kernel<<<cdiv((long long)kB * kS * 64, 256), 256, 0, stream>>>(imgF, imgT);

  // Image branch: img = relu(w_img @ x); ri = tanh(fc2); att; img_new -> fuseIn[96:192]
  conv(imgT, WpImg, bImg, img, nullptr, kB * kS, 64, 2, 6, 96, 96, 0, 0, 1, 0);
  conv(img, WpFc2, bFc2, ri, nullptr, kB * kS, 96, 3, 2, 24, 32, 0, 0, 2, 0);
  att_kernel<<<cdiv((long long)kB * kS, 256), 256, 0, stream>>>(ri, wFc3, bFc3, img, imgr);
  conv(imgr, WpPc, bPc, fuseIn, nullptr, kB * kS, 96, 3, 6, 96, 192, 96, 0, 1, 0);

  // Scale 0: r=0.5, K=16, MLP 9->16->16->32 -> maxpool -> fuseIn[0:32]
  ball_query_kernel<<<cdiv((long long)kB * kS * 32, 256), 256, 0, stream>>>(xyz, newXyz, idx0, 16, 0.25f);
  group_kernel<<<cdiv((long long)kB * kS * 16, 256), 256, 0, stream>>>(xyz, newXyz, feat, idx0, h0s0, 16);
  const int P0 = kB * kS * 16;
  conv(h0s0, Wp00, b00, a0f, nullptr, P0, 32, 1, 1, 16, 32, 0, 1, 1, 0);
  conv(a0f, Wp01, b01, b0f, nullptr, P0, 32, 1, 1, 16, 32, 0, 1, 1, 0);
  conv(b0f, Wp02, b02, h0s0, nullptr, P0, 32, 1, 2, 32, 32, 0, 0, 1, 0);
  pool_kernel<<<cdiv((long long)kB * kS * 32, 256), 256, 0, stream>>>(h0s0, fuseIn, 16, 32, 32, 0);

  // Scale 1: r=1.0, K=32, MLP 9->32->32->64 -> maxpool -> fuseIn[32:96]
  ball_query_kernel<<<cdiv((long long)kB * kS * 32, 256), 256, 0, stream>>>(xyz, newXyz, idx1, 32, 1.0f);
  group_kernel<<<cdiv((long long)kB * kS * 32, 256), 256, 0, stream>>>(xyz, newXyz, feat, idx1, h0s1, 32);
  const int P1 = kB * kS * 32;
  conv(h0s1, Wp10, b10, a1f, nullptr, P1, 32, 1, 2, 32, 32, 0, 0, 1, 0);
  conv(a1f, Wp11, b11, b1f, nullptr, P1, 32, 1, 2, 32, 32, 0, 0, 1, 0);
  conv(b1f, Wp12, b12, out1, nullptr, P1, 32, 1, 4, 64, 64, 0, 0, 1, 0);
  pool_kernel<<<cdiv((long long)kB * kS * 64, 256), 256, 0, stream>>>(out1, fuseIn, 32, 64, 64, 32);

  // Fuse: relu(w_fuse @ [point_feats; img_new]) -> d_out f32 (B,128,4096)
  conv(fuseIn, WpFuse, bFuse, nullptr, out, kB * kS, 192, 6, 8, 128, 0, 0, 0, 1, kS);
}
